// BertSelfAttention_55336358642447
// MI455X (gfx1250) — matrix operand order repaired and checked by reading, not compiled, not run on hardware
//
#include <hip/hip_runtime.h>

#define B_    4
#define S_    2048
#define HID_  768
#define NH_   12
#define HD_   64
#define M_    (B_ * S_)   // 8192 rows

typedef __bf16 bf16_t;
typedef bf16_t       v16bf __attribute__((ext_vector_type(16)));
typedef bf16_t       v8bf  __attribute__((ext_vector_type(8)));
typedef float        v8f   __attribute__((ext_vector_type(8)));
typedef unsigned int v8u32 __attribute__((ext_vector_type(8)));

__device__ __forceinline__ unsigned short bfbits(float f) {
    return __builtin_bit_cast(unsigned short, (bf16_t)f);
}
__device__ __forceinline__ unsigned packbf2(float a, float b) {
    return (unsigned)bfbits(a) | ((unsigned)bfbits(b) << 16);
}

// 16 contiguous bf16 (32B) -> v16bf   (B-fragment)
__device__ __forceinline__ v16bf loadbf16(const bf16_t* __restrict__ p) {
    uint4 a = ((const uint4*)p)[0];
    uint4 b = ((const uint4*)p)[1];
    v8u32 u;
    u[0] = a.x; u[1] = a.y; u[2] = a.z; u[3] = a.w;
    u[4] = b.x; u[5] = b.y; u[6] = b.z; u[7] = b.w;
    return __builtin_bit_cast(v16bf, u);
}

// two chunks of 8 contiguous bf16 (16B each) -> v16bf   (A-fragment)
__device__ __forceinline__ v16bf loadbf8x2(const bf16_t* __restrict__ p0,
                                           const bf16_t* __restrict__ p1) {
    uint4 a = *(const uint4*)p0;
    uint4 b = *(const uint4*)p1;
    v8u32 u;
    u[0] = a.x; u[1] = a.y; u[2] = a.z; u[3] = a.w;
    u[4] = b.x; u[5] = b.y; u[6] = b.z; u[7] = b.w;
    return __builtin_bit_cast(v16bf, u);
}

// ---------------------------------------------------------------------------
// Kernel 0: one-time fp32 -> bf16 conversion with optional scale
// (hardware v_cvt_pk_bf16_f32, bandwidth bound, runs once over ~32 MB of fp32 input)
// ---------------------------------------------------------------------------
__global__ __launch_bounds__(256) void cvt_bf16_kernel(
    const float* __restrict__ in, bf16_t* __restrict__ out, int n8, float scale) {
    int i = blockIdx.x * blockDim.x + threadIdx.x;
    if (i >= n8) return;
    float4 a = ((const float4*)in)[2 * i];
    float4 b = ((const float4*)in)[2 * i + 1];
    v8bf r;
    r[0] = (bf16_t)(a.x * scale); r[1] = (bf16_t)(a.y * scale);
    r[2] = (bf16_t)(a.z * scale); r[3] = (bf16_t)(a.w * scale);
    r[4] = (bf16_t)(b.x * scale); r[5] = (bf16_t)(b.y * scale);
    r[6] = (bf16_t)(b.z * scale); r[7] = (bf16_t)(b.w * scale);
    ((v8bf*)out)[i] = r;
}

// ---------------------------------------------------------------------------
// Kernel 1: y = hs @ W^T + bscale*bias (bf16 operands, fp32 accumulate).
//   transposed==0 -> out[b,h,s,d] bf16   (Q, K)
//   transposed==1 -> out[b,h,d,s] bf16   (V transposed; packed 16B stores)
// 256 threads = 8 waves; wave tile 32x32 (2x2 WMMA); block tile 128x64.
// ---------------------------------------------------------------------------
__global__ __launch_bounds__(256) void qkv_proj_kernel(
    const bf16_t* __restrict__ hsb, const bf16_t* __restrict__ Wb,
    const float* __restrict__ bias, bf16_t* __restrict__ out,
    int transposed, float bscale) {
    const int wave = threadIdx.x >> 5, lane = threadIdx.x & 31;
    const int half = lane >> 4, l16 = lane & 15;
    const int mbase = blockIdx.x * 128 + (wave & 3) * 32;
    const int nbase = blockIdx.y * 64 + (wave >> 2) * 32;

    v8f acc[2][2] = {};
    for (int kk = 0; kk < HID_; kk += 32) {
        v16bf a[2], b[2];
#pragma unroll
        for (int t = 0; t < 2; ++t) {
            const bf16_t* pa = hsb + (size_t)(mbase + t * 16 + l16) * HID_ + kk + half * 16;
            a[t] = loadbf8x2(pa, pa + 8);
            const int n = nbase + t * 16 + l16;
            b[t] = loadbf16(Wb + (size_t)n * HID_ + kk + half * 16);
        }
#pragma unroll
        for (int mt = 0; mt < 2; ++mt)
#pragma unroll
            for (int nt = 0; nt < 2; ++nt)
                acc[mt][nt] = __builtin_amdgcn_wmma_f32_16x16x32_bf16(
                    false, a[mt], false, b[nt], (short)0, acc[mt][nt], false, false);
    }

#pragma unroll
    for (int mt = 0; mt < 2; ++mt) {
#pragma unroll
        for (int nt = 0; nt < 2; ++nt) {
            const int n = nbase + nt * 16 + l16;   // C-layout: lane column
            const float bvn = bias[n] * bscale;
            const int h = n / HD_, d = n % HD_;
            if (!transposed) {
#pragma unroll
                for (int g = 0; g < 8; ++g) {      // row m = g + 8*half
                    const int m = mbase + mt * 16 + g + half * 8;
                    const int bb = m >> 11, s = m & (S_ - 1);
                    out[(((size_t)(bb * NH_ + h)) * S_ + s) * HD_ + d] =
                        (bf16_t)(acc[mt][nt][g] + bvn);
                }
            } else {
                // rows g are 8 consecutive s -> contiguous 16B packed store
                const int m0 = mbase + mt * 16 + half * 8;
                const int bb = m0 >> 11, s0 = m0 & (S_ - 1);
                const v8f& c = acc[mt][nt];
                uint4 st;
                st.x = packbf2(c[0] + bvn, c[1] + bvn);
                st.y = packbf2(c[2] + bvn, c[3] + bvn);
                st.z = packbf2(c[4] + bvn, c[5] + bvn);
                st.w = packbf2(c[6] + bvn, c[7] + bvn);
                *(uint4*)(out + (((size_t)(bb * NH_ + h)) * HD_ + d) * S_ + s0) = st;
            }
        }
    }
}

// ---------------------------------------------------------------------------
// Kernel 2: flash attention, bf16 operands, 64-key chunks.
// Each wave owns 16 query rows; 8 waves/block -> 128 q-rows/block.
// Per chunk: 8 WMMA (QK^T, scale pre-folded into Q) -> row-max via shfl_xor,
// exp -> P via LDS (C->A layout) -> row-sum via 2 WMMA against ones-B ->
// 8 WMMA (PV).
// ---------------------------------------------------------------------------
__global__ __launch_bounds__(256) void attn_kernel(
    const bf16_t* __restrict__ Q, const bf16_t* __restrict__ K,
    const bf16_t* __restrict__ Vt, const float* __restrict__ mask,
    float* __restrict__ out) {
    __shared__ __align__(16) unsigned int Plds[8 * 16 * 32];  // 8 waves x 16 rows x 64 bf16

    const int wave = threadIdx.x >> 5, lane = threadIdx.x & 31;
    const int half = lane >> 4, l16 = lane & 15;
    const int head = blockIdx.y;
    const int b = head / NH_, h = head % NH_;
    const bf16_t* Qh = Q + (size_t)head * S_ * HD_;
    const bf16_t* Kh = K + (size_t)head * S_ * HD_;
    const bf16_t* Vh = Vt + (size_t)head * HD_ * S_;
    const float* mk = mask + (size_t)b * S_;
    const int qbase = blockIdx.x * 128 + wave * 16;

    // Q fragments for d=0..31 and d=32..63 (1/sqrt(HD) already folded into Wq/bq)
    v16bf aq[2];
    {
        const bf16_t* p = Qh + (size_t)(qbase + l16) * HD_ + half * 16;
        aq[0] = loadbf8x2(p, p + 8);
        aq[1] = loadbf8x2(p + 32, p + 40);
    }

    // all-ones bf16 B-fragment for row-sum matmul
    v16bf vones;
    {
        v8u32 uo;
#pragma unroll
        for (int i = 0; i < 8; ++i) uo[i] = 0x3F803F80u;  // {1.0bf16, 1.0bf16}
        vones = __builtin_bit_cast(v16bf, uo);
    }

    float rm[8], rl[8];
#pragma unroll
    for (int g = 0; g < 8; ++g) { rm[g] = -3.0e38f; rl[g] = 0.0f; }
    v8f ctx[4] = {};

    bf16_t* myP = (bf16_t*)Plds + wave * 16 * 64;

    for (int kb = 0; kb < S_; kb += 64) {
        if (kb + 64 < S_)  // prefetch next K chunk (global_prefetch_b8)
            __builtin_prefetch(Kh + (size_t)(kb + 64 + l16) * HD_, 0, 1);

        // ---- scores: four 16(q) x 16(key) tiles over d = 0..63 ----
        v8f sc[4];
        float mv[4];
#pragma unroll
        for (int t = 0; t < 4; ++t) {
            const bf16_t* kp = Kh + (size_t)(kb + t * 16 + l16) * HD_;
            v8f c = {};
            c = __builtin_amdgcn_wmma_f32_16x16x32_bf16(
                false, aq[0], false, loadbf16(kp + half * 16), (short)0, c, false, false);
            c = __builtin_amdgcn_wmma_f32_16x16x32_bf16(
                false, aq[1], false, loadbf16(kp + 32 + half * 16), (short)0, c, false, false);
            mv[t] = mk[kb + t * 16 + l16];
            sc[t] = c;
        }

        // ---- row max + exp: row m = g + 8*half lives in 16 lanes of a half ----
        float alpha[8];
#pragma unroll
        for (int g = 0; g < 8; ++g) {
            float x0 = sc[0][g] + mv[0];
            float x1 = sc[1][g] + mv[1];
            float x2 = sc[2][g] + mv[2];
            float x3 = sc[3][g] + mv[3];
            float cm = fmaxf(fmaxf(x0, x1), fmaxf(x2, x3));
            cm = fmaxf(cm, __shfl_xor(cm, 1, 32));
            cm = fmaxf(cm, __shfl_xor(cm, 2, 32));
            cm = fmaxf(cm, __shfl_xor(cm, 4, 32));
            cm = fmaxf(cm, __shfl_xor(cm, 8, 32));
            const float nm = fmaxf(rm[g], cm);
            alpha[g] = __expf(rm[g] - nm);
            rm[g] = nm;
            sc[0][g] = __expf(x0 - nm);
            sc[1][g] = __expf(x1 - nm);
            sc[2][g] = __expf(x2 - nm);
            sc[3][g] = __expf(x3 - nm);
        }
#pragma unroll
        for (int nt = 0; nt < 4; ++nt)
#pragma unroll
            for (int g = 0; g < 8; ++g) ctx[nt][g] *= alpha[g];

        // ---- P: C-layout -> LDS bf16 -> two A-layout fragments (same-wave DS) ----
#pragma unroll
        for (int t = 0; t < 4; ++t)
#pragma unroll
            for (int g = 0; g < 8; ++g) {
                const int m = g + half * 8;
                myP[m * 64 + t * 16 + l16] = (bf16_t)sc[t][g];
            }
        v16bf ap[2];
#pragma unroll
        for (int j = 0; j < 2; ++j) {
            const unsigned int* r = Plds + (wave * 16 + l16) * 32;
            const int c = j * 16 + half * 8;
            v8u32 u;
            u[0] = r[c];     u[1] = r[c + 1]; u[2] = r[c + 2];  u[3] = r[c + 3];
            u[4] = r[c + 4]; u[5] = r[c + 5]; u[6] = r[c + 6]; u[7] = r[c + 7];
            ap[j] = __builtin_bit_cast(v16bf, u);
        }

        // ---- row sum = P x ones via WMMA (every C column equals the row sum) ----
        {
            v8f rsv = {};
            rsv = __builtin_amdgcn_wmma_f32_16x16x32_bf16(
                false, ap[0], false, vones, (short)0, rsv, false, false);
            rsv = __builtin_amdgcn_wmma_f32_16x16x32_bf16(
                false, ap[1], false, vones, (short)0, rsv, false, false);
#pragma unroll
            for (int g = 0; g < 8; ++g) rl[g] = rl[g] * alpha[g] + rsv[g];
        }

        // ---- PV: 4 output-d tiles x 2 k-steps, V^T rows contiguous over keys ----
#pragma unroll
        for (int nt = 0; nt < 4; ++nt) {
            const bf16_t* vp = Vh + (size_t)(nt * 16 + l16) * S_ + kb;
            ctx[nt] = __builtin_amdgcn_wmma_f32_16x16x32_bf16(
                false, ap[0], false, loadbf16(vp + half * 16), (short)0, ctx[nt], false, false);
            ctx[nt] = __builtin_amdgcn_wmma_f32_16x16x32_bf16(
                false, ap[1], false, loadbf16(vp + 32 + half * 16), (short)0, ctx[nt], false, false);
        }
    }

    // ---- epilogue: divide by row sum, scatter to [B,S,HID] merged-head f32 ----
#pragma unroll
    for (int g = 0; g < 8; ++g) {
        const float inv = (rl[g] > 0.0f) ? 1.0f / rl[g] : 0.0f;
        const int s = qbase + g + half * 8;
#pragma unroll
        for (int nt = 0; nt < 4; ++nt) {
            const int d = nt * 16 + l16;
            out[((size_t)(b * S_ + s)) * HID_ + h * HD_ + d] = ctx[nt][g] * inv;
        }
    }
}

extern "C" void kernel_launch(void* const* d_in, const int* in_sizes, int n_in,
                              void* d_out, int out_size, void* d_ws, size_t ws_size,
                              hipStream_t stream) {
    const float* hs   = (const float*)d_in[0];
    const float* mask = (const float*)d_in[1];
    const float* Wq   = (const float*)d_in[2];
    const float* bq   = (const float*)d_in[3];
    const float* Wk   = (const float*)d_in[4];
    const float* bk   = (const float*)d_in[5];
    const float* Wv   = (const float*)d_in[6];
    const float* bv   = (const float*)d_in[7];
    float* out = (float*)d_out;

    const size_t nHS = (size_t)M_ * HID_;     // 6,291,456
    const size_t nW  = (size_t)HID_ * HID_;   //   589,824
    bf16_t* hsb = (bf16_t*)d_ws;
    bf16_t* Wqb = hsb + nHS;
    bf16_t* Wkb = Wqb + nW;
    bf16_t* Wvb = Wkb + nW;
    bf16_t* Qb  = Wvb + nW;
    bf16_t* Kb  = Qb + nHS;
    bf16_t* Vtb = Kb + nHS;

    const float qscale = 0.125f;  // 1/sqrt(HD), folded into the Q projection
    const int h8 = (int)(nHS / 8), w8 = (int)(nW / 8);
    cvt_bf16_kernel<<<(h8 + 255) / 256, 256, 0, stream>>>(hs, hsb, h8, 1.0f);
    cvt_bf16_kernel<<<(w8 + 255) / 256, 256, 0, stream>>>(Wq, Wqb, w8, qscale);
    cvt_bf16_kernel<<<(w8 + 255) / 256, 256, 0, stream>>>(Wk, Wkb, w8, 1.0f);
    cvt_bf16_kernel<<<(w8 + 255) / 256, 256, 0, stream>>>(Wv, Wvb, w8, 1.0f);

    dim3 gp(M_ / 128, HID_ / 64);
    qkv_proj_kernel<<<gp, 256, 0, stream>>>(hsb, Wqb, bq, Qb, 0, qscale);
    qkv_proj_kernel<<<gp, 256, 0, stream>>>(hsb, Wkb, bk, Kb, 0, 1.0f);
    qkv_proj_kernel<<<gp, 256, 0, stream>>>(hsb, Wvb, bv, Vtb, 1, 1.0f);

    dim3 ga(S_ / 128, B_ * NH_);
    attn_kernel<<<ga, 256, 0, stream>>>(Qb, Kb, Vtb, mask, out);
}
